// Local_Attention_20495583936572
// MI455X (gfx1250) — matrix operand order repaired and checked by reading, not compiled, not run on hardware
//
#include <hip/hip_runtime.h>

// Problem constants (reference: B=2, T=2048, C=1024, H=16, D=64, causal mask)
#define B_ 2
#define T_ 2048
#define C_ 1024
#define H_ 16
#define D_ 64
#define SCALE_ 0.125f   // 1/sqrt(64)

typedef __attribute__((ext_vector_type(16))) __bf16        v16bf;
typedef __attribute__((ext_vector_type(8)))  float         v8f;
typedef __attribute__((ext_vector_type(4)))  unsigned int  u32x4;

union Frag {            // 16 bf16 per lane = 8 VGPRs (one WMMA A or B operand)
  v16bf v;
  u32x4 q[2];
};

__device__ __forceinline__ unsigned short f2bf(float f) {
  union { float f; unsigned int u; } c; c.f = f;
  unsigned int r = c.u + 0x7fffu + ((c.u >> 16) & 1u);   // round-to-nearest-even
  return (unsigned short)(r >> 16);
}

__device__ __forceinline__ v8f wmma_bf16(const Frag& a, const Frag& b, v8f c) {
  // (neg_a, A, neg_b, B, c_mod, C, reuse_a, reuse_b)
  return __builtin_amdgcn_wmma_f32_16x16x32_bf16(false, a.v, false, b.v,
                                                 (short)0, c, false, false);
}

// Async global->LDS 128-bit copy (VFLAT GV form), tracked by ASYNCcnt.
// The LDS destination VGPR holds the wave-relative LDS byte offset, which is
// the low 32 bits of the generic pointer.
__device__ __forceinline__ void async_ld_b128(const unsigned short* g,
                                              unsigned short* l) {
  unsigned lo = (unsigned)(unsigned long long)l;
  asm volatile("global_load_async_to_lds_b128 %0, %1, off"
               :: "v"(lo), "v"(g) : "memory");
}

// ---------------------------------------------------------------------------
// fp32 -> bf16 bulk convert (x)
// ---------------------------------------------------------------------------
__global__ __launch_bounds__(256)
void cvt_f32_bf16(const float* __restrict__ in, unsigned short* __restrict__ out) {
  int idx = (blockIdx.x * 256 + threadIdx.x) * 4;
  #pragma unroll
  for (int k = 0; k < 4; ++k) out[idx + k] = f2bf(in[idx + k]);
}

// ---------------------------------------------------------------------------
// Wt[k*C + n] = bf16(W[n*C + k])  (so B-matrix rows are contiguous: B[k][n]=W[n][k])
// ---------------------------------------------------------------------------
__global__ __launch_bounds__(256)
void transpose_cvt(const float* __restrict__ W, unsigned short* __restrict__ Wt) {
  int idx = blockIdx.x * 256 + threadIdx.x;   // over C*C, coalesced writes
  int n = idx & (C_ - 1);
  int k = idx >> 10;
  Wt[idx] = f2bf(W[n * C_ + k]);
}

// ---------------------------------------------------------------------------
// bf16 WMMA GEMM:  C[M,N] = A[M,K] * B[K,N]
//   block = 256 thr = 8 waves; block tile 128x128; wave tile 32x64 (2x4 WMMA)
// MODE 0: out bf16, scattered to [b][h][t][d]     (Q, V)
// MODE 1: out bf16, scattered to [b][h][d][t]     (K transposed)
// MODE 2: out f32, plain row-major M x N          (final projection)
// ---------------------------------------------------------------------------
template <int MODE>
__global__ __launch_bounds__(256)
void gemm_wmma(const unsigned short* __restrict__ A,
               const unsigned short* __restrict__ Bw,
               void* __restrict__ outp, int M, int N, int K) {
  const int lane = threadIdx.x & 31;
  const int w    = threadIdx.x >> 5;
  const int wm   = w & 3, wn = w >> 2;
  const int lg   = lane >> 4, l15 = lane & 15;
  const int bm   = blockIdx.x * 128 + wm * 32;
  const int bn   = blockIdx.y * 128 + wn * 64;

  v8f acc[2][4] = {};
  for (int k0 = 0; k0 < K; k0 += 32) {
    Frag a[2], b[4];
    #pragma unroll
    for (int i = 0; i < 2; ++i) {
      // 16-bit A 16x32 layout: lane l15 holds row M=l15; lanes>=16 take K base +8
      const unsigned short* p = A + (size_t)(bm + i * 16 + l15) * K + k0 + lg * 8;
      a[i].q[0] = *(const u32x4*)p;          // K = base .. base+7
      a[i].q[1] = *(const u32x4*)(p + 16);   // K = base+16 .. base+23
    }
    #pragma unroll
    for (int j = 0; j < 4; ++j) {
      // B 32x16 layout: lane holds column N=l15; K in same order as A
      #pragma unroll
      for (int s = 0; s < 16; ++s) {
        int kx = (s < 8) ? lg * 8 + s : 16 + lg * 8 + (s - 8);
        ((unsigned short*)&b[j])[s] = Bw[(size_t)(k0 + kx) * N + bn + j * 16 + l15];
      }
    }
    #pragma unroll
    for (int i = 0; i < 2; ++i)
      #pragma unroll
      for (int j = 0; j < 4; ++j)
        acc[i][j] = wmma_bf16(a[i], b[j], acc[i][j]);
  }

  // C/D layout: VGPR r, lane -> M = r + 8*(lane>>4), N = lane&15
  #pragma unroll
  for (int i = 0; i < 2; ++i)
    #pragma unroll
    for (int j = 0; j < 4; ++j)
      #pragma unroll
      for (int r = 0; r < 8; ++r) {
        int m = bm + i * 16 + r + 8 * lg;
        int n = bn + j * 16 + l15;
        float val = acc[i][j][r];
        if (MODE == 2) {
          ((float*)outp)[(size_t)m * N + n] = val;
        } else {
          int bb = m >> 11, t = m & (T_ - 1);   // rows = b*T + t
          int h  = n >> 6,  d = n & (D_ - 1);   // cols = h*64 + d
          if (MODE == 0)
            ((unsigned short*)outp)[(((size_t)(bb * H_ + h)) * T_ + t) * D_ + d] = f2bf(val);
          else
            ((unsigned short*)outp)[(((size_t)(bb * H_ + h)) * D_ + d) * T_ + t] = f2bf(val);
        }
      }
}

// ---------------------------------------------------------------------------
// Causal flash attention, bf16 WMMA, fp32 online-softmax accumulators.
// grid = (B*H, T/256), block = 256 (8 waves); each wave owns 32 query rows.
// V tiles are staged global->LDS with async copies (ASYNCcnt) issued before
// the QK^T WMMAs + softmax, hiding the fetch latency.
// Q : [bh][t][d], Kt : [bh][d][t], V : [bh][t][d]; out : [b][t][h*64+d] bf16
// ---------------------------------------------------------------------------
__global__ __launch_bounds__(256)
void flash_attn(const unsigned short* __restrict__ Q,
                const unsigned short* __restrict__ Kt,
                const unsigned short* __restrict__ V,
                unsigned short* __restrict__ O) {
  __shared__ alignas(16) unsigned short pbuf[8][32][32];   // per-wave P staging
  __shared__ alignas(16) unsigned short vbuf[8][32 * 64];  // per-wave V tile
  const int lane = threadIdx.x & 31;
  const int w    = threadIdx.x >> 5;
  const int lg   = lane >> 4, l15 = lane & 15;
  const int bh   = blockIdx.x;
  const int bb   = bh >> 4, h = bh & 15;
  const int q0   = blockIdx.y * 256 + w * 32;   // wave's first query row

  const unsigned short* Qb  = Q  + (size_t)bh * T_ * D_;
  const unsigned short* Ktb = Kt + (size_t)bh * D_ * T_;
  const unsigned short* Vb  = V  + (size_t)bh * T_ * D_;

  // Q fragments: 2 M-tiles x 2 K(d)-steps, resident whole kernel
  Frag qa[2][2];
  #pragma unroll
  for (int i = 0; i < 2; ++i)
    #pragma unroll
    for (int kk = 0; kk < 2; ++kk) {
      const unsigned short* p = Qb + (size_t)(q0 + i * 16 + l15) * D_ + kk * 32 + lg * 8;
      qa[i][kk].q[0] = *(const u32x4*)p;
      qa[i][kk].q[1] = *(const u32x4*)(p + 16);
    }

  v8f o[2][4] = {};
  float mrow[2][8], lrow[2][8];
  #pragma unroll
  for (int i = 0; i < 2; ++i)
    #pragma unroll
    for (int r = 0; r < 8; ++r) { mrow[i][r] = -3.0e38f; lrow[i][r] = 0.0f; }

  for (int kb = 0; kb <= q0; kb += 32) {
    // WAR: previous iteration's DS reads of vbuf must retire before the async
    // engine overwrites it (async LDS writes are unordered vs DS ops).
    asm volatile("s_wait_dscnt 0x0" ::: "memory");
    // Stage this block's V tile (keys kb..kb+31, all 64 d) into LDS: each lane
    // copies its own key row in 8x16B async transfers.
    #pragma unroll
    for (int c = 0; c < 8; ++c)
      async_ld_b128(Vb + (size_t)(kb + lane) * D_ + c * 8,
                    &vbuf[w][lane * D_ + c * 8]);
    if (kb + 32 <= q0)   // pull next K block toward the caches
      __builtin_prefetch(Ktb + (size_t)lane * T_ + kb + 32, 0, 1);

    // S = Q * K^T : B-matrix rows are d, contiguous over keys in Kt
    Frag kf[2][2];
    #pragma unroll
    for (int kk = 0; kk < 2; ++kk)
      #pragma unroll
      for (int j = 0; j < 2; ++j) {
        #pragma unroll
        for (int s = 0; s < 16; ++s) {
          int kx = (s < 8) ? lg * 8 + s : 16 + lg * 8 + (s - 8);
          ((unsigned short*)&kf[kk][j])[s] = Ktb[(size_t)(kk * 32 + kx) * T_ + kb + j * 16 + l15];
        }
      }
    v8f s[2][2] = {};
    #pragma unroll
    for (int i = 0; i < 2; ++i)
      #pragma unroll
      for (int j = 0; j < 2; ++j) {
        s[i][j] = wmma_bf16(qa[i][0], kf[0][j], s[i][j]);
        s[i][j] = wmma_bf16(qa[i][1], kf[1][j], s[i][j]);
      }

    const bool diag = (kb == q0);   // only the diagonal block needs masking
    #pragma unroll
    for (int i = 0; i < 2; ++i) {
      float alpha[8];
      #pragma unroll
      for (int r = 0; r < 8; ++r) {
        int qr = i * 16 + r + 8 * lg;              // row within wave tile
        float s0 = s[i][0][r] * SCALE_;
        float s1 = s[i][1][r] * SCALE_;
        if (diag) {                                 // key > query  =>  -inf
          if (l15 > qr)      s0 = -3.0e38f;
          if (16 + l15 > qr) s1 = -3.0e38f;
        }
        // row max across the 16 lanes holding this row (xor<16 stays in half)
        float pm = fmaxf(s0, s1);
        pm = fmaxf(pm, __shfl_xor(pm, 1, 32));
        pm = fmaxf(pm, __shfl_xor(pm, 2, 32));
        pm = fmaxf(pm, __shfl_xor(pm, 4, 32));
        pm = fmaxf(pm, __shfl_xor(pm, 8, 32));
        float mn = fmaxf(mrow[i][r], pm);
        float al = __expf(mrow[i][r] - mn);
        mrow[i][r] = mn;
        float p0 = __expf(s0 - mn);
        float p1 = __expf(s1 - mn);
        float ps = p0 + p1;
        ps += __shfl_xor(ps, 1, 32);
        ps += __shfl_xor(ps, 2, 32);
        ps += __shfl_xor(ps, 4, 32);
        ps += __shfl_xor(ps, 8, 32);
        lrow[i][r] = lrow[i][r] * al + ps;
        alpha[r] = al;
        // stage P (C/D layout -> LDS row-major) for re-read in A layout
        pbuf[w][i * 16 + r + 8 * lg][l15]      = f2bf(p0);
        pbuf[w][i * 16 + r + 8 * lg][16 + l15] = f2bf(p1);
      }
      #pragma unroll
      for (int j = 0; j < 4; ++j)
        #pragma unroll
        for (int r = 0; r < 8; ++r)
          o[i][j][r] *= alpha[r];
    }

    // same-wave DS ops complete in order; wait for the staged P stores,
    // and for the async V tile to land in LDS.
    asm volatile("s_wait_dscnt 0x0" ::: "memory");
    asm volatile("s_wait_asynccnt 0x0" ::: "memory");

    Frag pa[2];
    #pragma unroll
    for (int i = 0; i < 2; ++i) {  // re-read P as a 16x32 A fragment
      pa[i].q[0] = *(const u32x4*)&pbuf[w][i * 16 + l15][lg * 8];
      pa[i].q[1] = *(const u32x4*)&pbuf[w][i * 16 + l15][lg * 8 + 16];
    }
    Frag vf[4];
    #pragma unroll
    for (int j = 0; j < 4; ++j) {  // V from LDS: lane = d column, keys in A's k order
      #pragma unroll
      for (int s = 0; s < 16; ++s) {
        int kx = (s < 8) ? lg * 8 + s : 16 + lg * 8 + (s - 8);
        ((unsigned short*)&vf[j])[s] = vbuf[w][kx * D_ + j * 16 + l15];
      }
    }
    #pragma unroll
    for (int i = 0; i < 2; ++i)
      #pragma unroll
      for (int j = 0; j < 4; ++j)
        o[i][j] = wmma_bf16(pa[i], vf[j], o[i][j]);

    asm volatile("" ::: "memory");  // keep next-iter LDS stores after these reads
  }

  // normalize and write [b][t][h*64+d] bf16 for the final projection GEMM
  #pragma unroll
  for (int i = 0; i < 2; ++i)
    #pragma unroll
    for (int r = 0; r < 8; ++r) {
      float inv = 1.0f / lrow[i][r];
      int qg = q0 + i * 16 + r + 8 * lg;
      size_t base = ((size_t)bb * T_ + qg) * C_ + h * D_;
      #pragma unroll
      for (int j = 0; j < 4; ++j)
        O[base + j * 16 + l15] = f2bf(o[i][j][r] * inv);
    }
}

// ---------------------------------------------------------------------------
extern "C" void kernel_launch(void* const* d_in, const int* in_sizes, int n_in,
                              void* d_out, int out_size, void* d_ws, size_t ws_size,
                              hipStream_t stream) {
  const float* x  = (const float*)d_in[0];
  const float* Wq = (const float*)d_in[1];
  const float* Wk = (const float*)d_in[2];
  const float* Wv = (const float*)d_in[3];
  const float* Wo = (const float*)d_in[4];
  float* out = (float*)d_out;

  const size_t XN = (size_t)B_ * T_ * C_;   // 4M elems
  const size_t WN = (size_t)C_ * C_;        // 1M elems
  unsigned short* ws  = (unsigned short*)d_ws;
  unsigned short* xb  = ws;          // x in bf16
  unsigned short* wqT = xb + XN;
  unsigned short* wkT = wqT + WN;
  unsigned short* wvT = wkT + WN;
  unsigned short* woT = wvT + WN;
  unsigned short* Qb  = woT + WN;    // [b][h][t][d]
  unsigned short* Ktb = Qb + XN;     // [b][h][d][t]
  unsigned short* Vb  = Ktb + XN;    // [b][h][t][d]
  unsigned short* Ab  = Vb + XN;     // attention out, [b][t][h*64+d]

  cvt_f32_bf16<<<dim3((unsigned)(XN / 1024)), dim3(256), 0, stream>>>(x, xb);
  transpose_cvt<<<dim3((unsigned)(WN / 256)), dim3(256), 0, stream>>>(Wq, wqT);
  transpose_cvt<<<dim3((unsigned)(WN / 256)), dim3(256), 0, stream>>>(Wk, wkT);
  transpose_cvt<<<dim3((unsigned)(WN / 256)), dim3(256), 0, stream>>>(Wv, wvT);
  transpose_cvt<<<dim3((unsigned)(WN / 256)), dim3(256), 0, stream>>>(Wo, woT);

  dim3 gg(4096 / 128, 1024 / 128);   // (32, 8)
  gemm_wmma<0><<<gg, dim3(256), 0, stream>>>(xb, wqT, Qb, 4096, 1024, 1024);
  gemm_wmma<1><<<gg, dim3(256), 0, stream>>>(xb, wkT, Ktb, 4096, 1024, 1024);
  gemm_wmma<0><<<gg, dim3(256), 0, stream>>>(xb, wvT, Vb, 4096, 1024, 1024);

  flash_attn<<<dim3(B_ * H_, T_ / 256), dim3(256), 0, stream>>>(Qb, Ktb, Vb, Ab);

  gemm_wmma<2><<<gg, dim3(256), 0, stream>>>(Ab, woT, out, 4096, 1024, 1024);
}
